// SfaRNN_10222022164700
// MI455X (gfx1250) — matrix operand order repaired and checked by reading, not compiled, not run on hardware
//
#include <hip/hip_runtime.h>

// ---------------------------------------------------------------------------
// SFA-RNN fused step for MI455X (gfx1250, wave32, WMMA)
//   v_new = relu((1-1/150)*v + (10/150)*u)
//   drive = x@win^T + u@wr^T + bias - v_new   (fused GEMM, K = 1024+2048)
//   u_new = relu(0.9*u + 0.1*drive)
// Split-bf16 (hi + lo residual) emulated-fp32 GEMM on v_wmma_f32_16x16x32_bf16.
// Conversion done ONCE at LDS staging; fragments are pure ds_load_b128 of bf16.
// ---------------------------------------------------------------------------

typedef __attribute__((ext_vector_type(16))) __bf16 v16bf;
typedef __attribute__((ext_vector_type(8)))  __bf16 v8bf;
typedef __attribute__((ext_vector_type(4)))  __bf16 v4bf;
typedef __attribute__((ext_vector_type(8)))  float  v8f;
typedef __attribute__((ext_vector_type(4)))  float  f4;

#define B_DIM 8192
#define D_IN  1024
#define H_DIM 2048
#define K_TOT (D_IN + H_DIM)   // 3072
#define LDB   40               // bf16 LDS row stride: 80B -> conflict-free b128 frag loads

union F16 { v16bf v; v8bf h[2]; };

__global__ __launch_bounds__(256) void sfa_rnn_step(
    const float* __restrict__ x,   const float* __restrict__ u_in,
    const float* __restrict__ v_in,const float* __restrict__ win,
    const float* __restrict__ wr,  const float* __restrict__ bias,
    float* __restrict__ u_out,     float* __restrict__ v_out)
{
  // 4 x (128 rows x 32 K, stride 40) bf16 tiles = 4 x 10 KB = 40 KB LDS
  __shared__ __bf16 lAh[128 * LDB];
  __shared__ __bf16 lAl[128 * LDB];
  __shared__ __bf16 lBh[128 * LDB];
  __shared__ __bf16 lBl[128 * LDB];

  const int tid  = threadIdx.x;
  const int wave = tid >> 5;       // 0..7  (8 wave32 per block)
  const int lane = tid & 31;
  const int lh   = lane >> 4;      // lane half: 0 or 1
  const int l15  = lane & 15;
  const int bm   = blockIdx.x;     // 64 M-blocks of 128 rows
  const int bn   = blockIdx.y;     // 16 N-blocks of 128 cols

  // ---- global -> register fetch of next K-slab (fp32) -----------------------
  auto stage_fetch = [&](int kk, f4* ra, f4* rb) {
    const float* aSrc; const float* bSrc; int ld; int kof;
    if (kk < D_IN) { aSrc = x;    bSrc = win; ld = D_IN;  kof = kk;        }
    else           { aSrc = u_in; bSrc = wr;  ld = H_DIM; kof = kk - D_IN; }
    #pragma unroll
    for (int i = 0; i < 4; ++i) {
      int flat = tid + i * 256;          // 1024 float4s per 128x32 tile
      int row  = flat >> 3;              // 0..127
      int c4   = (flat & 7) * 4;         // 0,4,...,28
      ra[i] = *(const f4*)(aSrc + (size_t)(bm * 128 + row) * ld + kof + c4);
      rb[i] = *(const f4*)(bSrc + (size_t)(bn * 128 + row) * ld + kof + c4);
    }
  };

  // fp32x4 -> (hi bf16x4, lo bf16x4) split: a ~= hi + lo, |lo| <= ulp(hi)/2
  auto cvt_split = [](const f4 f, v4bf& hi, v4bf& lo) {
    #pragma unroll
    for (int j = 0; j < 4; ++j) {
      __bf16 h = (__bf16)f[j];
      hi[j] = h;
      lo[j] = (__bf16)(f[j] - (float)h);
    }
  };

  // ---- register -> LDS store, converting ONCE per element -------------------
  auto stage_store = [&](const f4* ra, const f4* rb) {
    #pragma unroll
    for (int i = 0; i < 4; ++i) {
      int flat = tid + i * 256;
      int row  = flat >> 3;
      int c4   = (flat & 7) * 4;
      v4bf h, l;
      cvt_split(ra[i], h, l);
      *(v4bf*)(lAh + row * LDB + c4) = h;
      *(v4bf*)(lAl + row * LDB + c4) = l;
      cvt_split(rb[i], h, l);
      *(v4bf*)(lBh + row * LDB + c4) = h;
      *(v4bf*)(lBl + row * LDB + c4) = l;
    }
  };

  // ---- accumulators: 16(M) x 128(N) per wave = 8 WMMA tiles -----------------
  v8f acc[8];
  #pragma unroll
  for (int t = 0; t < 8; ++t)
    #pragma unroll
    for (int i = 0; i < 8; ++i) acc[t][i] = 0.0f;

  f4 ra[4], rb[4];
  stage_fetch(0, ra, rb);

  for (int kk = 0; kk < K_TOT; kk += 32) {
    __syncthreads();                 // prior compute done reading LDS
    stage_store(ra, rb);
    __syncthreads();
    if (kk + 32 < K_TOT) stage_fetch(kk + 32, ra, rb);  // overlap with compute

    // A fragment (16x32, ISA layout): lane l: M = l&15,
    //   lanes 0-15:  e0-7 -> K=0..7,   e8-15 -> K=16..23
    //   lanes 16-31: e0-7 -> K=8..15,  e8-15 -> K=24..31
    const int aOff = (wave * 16 + l15) * LDB + lh * 8;
    F16 ah, al;
    ah.h[0] = *(const v8bf*)(lAh + aOff);
    ah.h[1] = *(const v8bf*)(lAh + aOff + 16);
    al.h[0] = *(const v8bf*)(lAl + aOff);
    al.h[1] = *(const v8bf*)(lAl + aOff + 16);

    #pragma unroll
    for (int t = 0; t < 8; ++t) {
      // B fragment (32x16): lane l: N = l&15, lanes 0-15 K=0..7,16..23; 16-31 K=8..15,24..31
      const int bOff = (t * 16 + l15) * LDB + lh * 8;
      F16 bh, bl;
      bh.h[0] = *(const v8bf*)(lBh + bOff);
      bh.h[1] = *(const v8bf*)(lBh + bOff + 16);
      bl.h[0] = *(const v8bf*)(lBl + bOff);
      bl.h[1] = *(const v8bf*)(lBl + bOff + 16);

      // split-bf16 fp32 emulation: hi*hi + hi*lo + lo*hi
      acc[t] = __builtin_amdgcn_wmma_f32_16x16x32_bf16(
                   false, ah.v, false, bh.v, (short)0, acc[t], false, false);
      acc[t] = __builtin_amdgcn_wmma_f32_16x16x32_bf16(
                   false, ah.v, false, bl.v, (short)0, acc[t], false, false);
      acc[t] = __builtin_amdgcn_wmma_f32_16x16x32_bf16(
                   false, al.v, false, bh.v, (short)0, acc[t], false, false);
    }
  }

  // ---- fused elementwise epilogue ------------------------------------------
  // C/D layout: vgpr r, lane l -> M = r + (l>>4)*8, N = l&15
  const float c_vv = 1.0f - 1.0f / 150.0f;   // (1 - ALPHA_V)
  const float c_vu = 10.0f / 150.0f;         // ALPHA_V * M
  #pragma unroll
  for (int t = 0; t < 8; ++t) {
    #pragma unroll
    for (int r = 0; r < 8; ++r) {
      int gm = bm * 128 + wave * 16 + r + lh * 8;
      int gn = bn * 128 + t * 16 + l15;
      size_t idx = (size_t)gm * H_DIM + gn;
      float uv = u_in[idx];
      float vv = v_in[idx];
      float vn = c_vv * vv + c_vu * uv;
      vn = vn > 0.0f ? vn : 0.0f;                         // v_new
      float drive = acc[t][r] + bias[gn] - vn;
      float un = 0.9f * uv + 0.1f * drive;
      un = un > 0.0f ? un : 0.0f;                         // u_new
      u_out[idx] = un;
      v_out[idx] = vn;
    }
  }
}

extern "C" void kernel_launch(void* const* d_in, const int* in_sizes, int n_in,
                              void* d_out, int out_size, void* d_ws, size_t ws_size,
                              hipStream_t stream) {
  const float* x    = (const float*)d_in[0];
  const float* u    = (const float*)d_in[1];
  const float* v    = (const float*)d_in[2];
  const float* win  = (const float*)d_in[3];
  const float* wr   = (const float*)d_in[4];
  const float* bias = (const float*)d_in[5];
  float* u_out = (float*)d_out;                              // outputs concat:
  float* v_out = (float*)d_out + (size_t)B_DIM * H_DIM;      // (u_new, v_new)

  dim3 grid(B_DIM / 128, H_DIM / 128);   // 64 x 16 blocks
  sfa_rnn_step<<<grid, 256, 0, stream>>>(x, u, v, win, wr, bias, u_out, v_out);
}
